// MLPPredictor_33028298506368
// MI455X (gfx1250) — matrix operand order repaired and checked by reading, not compiled, not run on hardware
//
#include <hip/hip_runtime.h>
#include <hip/hip_bf16.h>

typedef __attribute__((ext_vector_type(16))) __bf16 bf16x16;
typedef __attribute__((ext_vector_type(8)))  __bf16 bf16x8;
typedef __attribute__((ext_vector_type(8)))  float  f32x8;

#define HFEAT 128
#define KCAT  256
#define EDGES_PER_BLOCK 64
#define THREADS 256
#define HE_STRIDE 264   // 256 bf16 + 8 pad -> +16B per row (bank rotate)
#define X1_STRIDE 136   // 128 bf16 + 8 pad

__device__ __forceinline__ unsigned short f2bf_bits(float f) {
  unsigned int u = __float_as_uint(f);
  u += 0x7FFFu + ((u >> 16) & 1u);     // round-to-nearest-even
  return (unsigned short)(u >> 16);
}

// One-time f32 -> bf16 conversion of W1 (128x256) and W2 (128x128) into d_ws.
__global__ void convert_weights(const float* __restrict__ W1,
                                const float* __restrict__ W2,
                                unsigned short* __restrict__ W1bf,
                                unsigned short* __restrict__ W2bf) {
  int i = blockIdx.x * blockDim.x + threadIdx.x;
  if (i < HFEAT * KCAT) W1bf[i] = f2bf_bits(W1[i]);
  int j = i - HFEAT * KCAT;
  if (j >= 0 && j < HFEAT * HFEAT) W2bf[j] = f2bf_bits(W2[j]);
}

__device__ __forceinline__ bf16x16 make_frag(const unsigned short* p) {
  bf16x8 lo = *(const bf16x8*)p;        // 16B -> ds/global b128
  bf16x8 hi = *(const bf16x8*)(p + 16);
  bf16x16 a;
#pragma unroll
  for (int i = 0; i < 8; ++i) { a[i] = lo[i]; a[i + 8] = hi[i]; }
  return a;
}

__global__ __launch_bounds__(THREADS)
void edge_mlp_kernel(const float* __restrict__ h,
                     const int*   __restrict__ src_idx,
                     const int*   __restrict__ dst_idx,
                     const unsigned short* __restrict__ W1bf,
                     const float* __restrict__ b1,
                     const unsigned short* __restrict__ W2bf,
                     const float* __restrict__ b2,
                     const float* __restrict__ W3,
                     const float* __restrict__ b3,
                     float* __restrict__ out,
                     int n_edges) {
  __shared__ unsigned short he_s[EDGES_PER_BLOCK * HE_STRIDE]; // 33792 B
  __shared__ unsigned short x1_s[EDGES_PER_BLOCK * X1_STRIDE]; // 17408 B
  __shared__ float part_s[8 * EDGES_PER_BLOCK];                //  2048 B

  const int tid   = threadIdx.x;
  const int wave  = tid >> 5;
  const int lane  = tid & 31;
  const int l16   = lane & 15;
  const int lhalf = lane >> 4;
  const long long ebase = (long long)blockIdx.x * EDGES_PER_BLOCK;

  // ---- Gather src/dst node rows (f32, L2-resident) -> bf16 in LDS --------
  const float4* hv = (const float4*)h;
#pragma unroll
  for (int it = 0; it < 16; ++it) {
    int flat = it * THREADS + tid;   // 0..4095 = 64 edges * 2 halves * 32 float4
    int e    = flat >> 6;
    int r    = flat & 63;
    int half = r >> 5;
    int f4   = r & 31;
    long long ge = ebase + e;
    if (ge >= n_edges) ge = n_edges - 1;   // clamp (duplicate compute, no write)
    int node = half ? dst_idx[ge] : src_idx[ge];
    float4 v = hv[(long long)node * 32 + f4];
    unsigned int p0 = (unsigned int)f2bf_bits(v.x) | ((unsigned int)f2bf_bits(v.y) << 16);
    unsigned int p1 = (unsigned int)f2bf_bits(v.z) | ((unsigned int)f2bf_bits(v.w) << 16);
    unsigned int* dst = (unsigned int*)&he_s[e * HE_STRIDE + half * HFEAT + f4 * 4];
    dst[0] = p0; dst[1] = p1;
  }
  __syncthreads();

  const int nrow = wave * 16 + l16;  // output feature column for this lane

  // ---- Layer 1: [64 x 256] @ W1^T -> [64 x 128], bf16 WMMA, f32 acc ------
  f32x8 acc[4];
#pragma unroll
  for (int mt = 0; mt < 4; ++mt)
#pragma unroll
    for (int j = 0; j < 8; ++j) acc[mt][j] = 0.0f;

#pragma unroll
  for (int kt = 0; kt < 8; ++kt) {
    const int k0 = kt * 32;
    // B: W1 row n, two 16B loads (k = 8h..8h+7, 16+8h..), reused over 4 m-tiles
    bf16x16 B = make_frag(W1bf + nrow * KCAT + k0 + lhalf * 8);
#pragma unroll
    for (int mt = 0; mt < 4; ++mt) {
      bf16x16 A = make_frag(&he_s[(mt * 16 + l16) * HE_STRIDE + k0 + lhalf * 8]);
      acc[mt] = __builtin_amdgcn_wmma_f32_16x16x32_bf16(
          false, A, false, B, (short)0, acc[mt], false, false);
    }
  }

  {
    float b1v = b1[nrow];
#pragma unroll
    for (int mt = 0; mt < 4; ++mt)
#pragma unroll
      for (int j = 0; j < 8; ++j) {
        float v = fmaxf(acc[mt][j] + b1v, 0.0f);
        int row = mt * 16 + lhalf * 8 + j;
        x1_s[row * X1_STRIDE + nrow] = f2bf_bits(v);
      }
  }
  __syncthreads();

  // ---- Layer 2: [64 x 128] @ W2^T -> [64 x 128] --------------------------
  f32x8 acc2[4];
#pragma unroll
  for (int mt = 0; mt < 4; ++mt)
#pragma unroll
    for (int j = 0; j < 8; ++j) acc2[mt][j] = 0.0f;

#pragma unroll
  for (int kt = 0; kt < 4; ++kt) {
    const int k0 = kt * 32;
    bf16x16 B = make_frag(W2bf + nrow * HFEAT + k0 + lhalf * 8);
#pragma unroll
    for (int mt = 0; mt < 4; ++mt) {
      bf16x16 A = make_frag(&x1_s[(mt * 16 + l16) * X1_STRIDE + k0 + lhalf * 8]);
      acc2[mt] = __builtin_amdgcn_wmma_f32_16x16x32_bf16(
          false, A, false, B, (short)0, acc2[mt], false, false);
    }
  }

  // ---- Layer 3: bias+relu, scale by w3[n], reduce over 128 features ------
  {
    float b2v = b2[nrow];
    float w3v = W3[nrow];
#pragma unroll
    for (int mt = 0; mt < 4; ++mt)
#pragma unroll
      for (int j = 0; j < 8; ++j) {
        float v = fmaxf(acc2[mt][j] + b2v, 0.0f) * w3v;
        v += __shfl_xor(v, 1, 16);
        v += __shfl_xor(v, 2, 16);
        v += __shfl_xor(v, 4, 16);
        v += __shfl_xor(v, 8, 16);
        if (l16 == 0)
          part_s[wave * EDGES_PER_BLOCK + mt * 16 + lhalf * 8 + j] = v;
      }
  }
  __syncthreads();

  if (tid < EDGES_PER_BLOCK) {
    long long ge = ebase + tid;
    if (ge < n_edges) {
      float s = b3[0];
#pragma unroll
      for (int w = 0; w < 8; ++w) s += part_s[w * EDGES_PER_BLOCK + tid];
      out[ge] = s;
    }
  }
}

extern "C" void kernel_launch(void* const* d_in, const int* in_sizes, int n_in,
                              void* d_out, int out_size, void* d_ws, size_t ws_size,
                              hipStream_t stream) {
  const float* h  = (const float*)d_in[0];
  const int* src  = (const int*)d_in[1];
  const int* dst  = (const int*)d_in[2];
  const float* W1 = (const float*)d_in[3];
  const float* b1 = (const float*)d_in[4];
  const float* W2 = (const float*)d_in[5];
  const float* b2 = (const float*)d_in[6];
  const float* W3 = (const float*)d_in[7];
  const float* b3 = (const float*)d_in[8];
  float* out = (float*)d_out;
  const int n_edges = in_sizes[1];

  unsigned short* W1bf = (unsigned short*)d_ws;              // 64 KB
  unsigned short* W2bf = W1bf + HFEAT * KCAT;                // 32 KB

  const int conv_elems = HFEAT * KCAT + HFEAT * HFEAT;       // 49152
  convert_weights<<<(conv_elems + 255) / 256, 256, 0, stream>>>(W1, W2, W1bf, W2bf);

  const int blocks = (n_edges + EDGES_PER_BLOCK - 1) / EDGES_PER_BLOCK;
  edge_mlp_kernel<<<blocks, THREADS, 0, stream>>>(
      h, src, dst, W1bf, b1, W2bf, b2, W3, b3, out, n_edges);
}
